// DynamicRNNEncoder_73547019976928
// MI455X (gfx1250) — matrix operand order repaired and checked by reading, not compiled, not run on hardware
//
#include <hip/hip_runtime.h>
#include <hip/hip_bf16.h>

// Problem dims (fixed by the reference)
#define BB 64
#define TT 128
#define DD 256
#define HH 256
#define H3 768   // 3*H
#define KC 96    // rows of R pinned in LDS (= 3 k-chunks of 32)

typedef __attribute__((ext_vector_type(16))) _Float16 v16h;
typedef __attribute__((ext_vector_type(8)))  _Float16 v8h;
typedef __attribute__((ext_vector_type(8)))  float    v8f;

// ---------------------------------------------------------------------------
// Kernel 1: weight precision conversion.
//   Kt : f16, transposed [3H][D]  (WMMA B-fragments load contiguous 32B)
//   Rh : f16, row-major  [H][3H]  (matvec loads contiguous 16B per thread)
// ---------------------------------------------------------------------------
__global__ void convert_weights(const float* __restrict__ Kw,
                                const float* __restrict__ Rw,
                                _Float16* __restrict__ Kt,
                                _Float16* __restrict__ Rh) {
    int idx = blockIdx.x * blockDim.x + threadIdx.x;
    if (idx < DD * H3) {
        int n = idx / DD;         // 0..767
        int k = idx % DD;         // 0..255
        Kt[idx] = (_Float16)Kw[k * H3 + n];   // transpose
        Rh[idx] = (_Float16)Rw[idx];          // straight copy
    }
}

// ---------------------------------------------------------------------------
// Kernel 2: MX = X[B*T, D] @ kernel[D, 3H] + bias0 via v_wmma_f32_16x16x32_f16.
// One wave per 16x64 strip: 4 N-tiles share one A-fragment (4x less X traffic,
// back-to-back WMMAs). Fragments use contiguous k = 16*hi + s in both A and B.
// ---------------------------------------------------------------------------
__global__ void mx_gemm_wmma(const float* __restrict__ X,       // [B*T, D] f32
                             const _Float16* __restrict__ Kt,   // [3H][D] f16
                             const float* __restrict__ bias0,   // [3H]
                             float* __restrict__ MX) {          // [B*T, 3H] f32
    const int NG = H3 / 64;                                     // 12 groups of 4 tiles
    const int wave = (blockIdx.x * blockDim.x + threadIdx.x) >> 5;
    const int lane = threadIdx.x & 31;
    const int mt = wave / NG;                                   // 0..511
    const int ng = wave % NG;                                   // 0..11

    const int lo = lane & 15;
    const int hi = lane >> 4;

    v8f acc0 = {0,0,0,0,0,0,0,0}, acc1 = acc0, acc2 = acc0, acc3 = acc0;

    const int arow = mt * 16 + lo;
    const int ncol = ng * 64 + lo;                              // tile-u col = ncol + 16u

    #pragma unroll
    for (int kk = 0; kk < DD / 32; ++kk) {
        // A fragment: lanes 0-15 rows M=lo, hi selects K-halves (contiguous k, same order as B)
        const float* xr = X + arow * DD + kk * 32 + 16 * hi;
        v16h a;
        #pragma unroll
        for (int e = 0; e < 8; ++e)  a[e]     = (_Float16)xr[e];        // K = 16*hi+e
        #pragma unroll
        for (int e = 0; e < 8; ++e)  a[8 + e] = (_Float16)xr[8 + e];    // K = 16*hi+8+e

        // 4 B fragments: 16 contiguous f16 along K of transposed kernel (32B each)
        const _Float16* bp = Kt + ncol * DD + kk * 32 + hi * 16;
        v8f* accs[4] = {&acc0, &acc1, &acc2, &acc3};
        #pragma unroll
        for (int u = 0; u < 4; ++u) {
            v16h b = *(const v16h*)(bp + u * 16 * DD);
            *accs[u] = __builtin_amdgcn_wmma_f32_16x16x32_f16(
                false, a, false, b, (short)0, *accs[u], false, false);
        }
    }

    v8f* accs[4] = {&acc0, &acc1, &acc2, &acc3};
    #pragma unroll
    for (int u = 0; u < 4; ++u) {
        const int col = ncol + u * 16;
        const float bv = bias0[col];
        #pragma unroll
        for (int v = 0; v < 8; ++v)                    // VGPR v -> M = v + 8*hi
            MX[(mt * 16 + 8 * hi + v) * H3 + col] = (*accs[u])[v] + bv;
    }
}

// ---------------------------------------------------------------------------
// Kernel 3: persistent recurrence; one WG per batch element (sequences are
// independent -> no grid sync). CDNA5's 320 KB LDS holds BOTH the full
// history Y_b (128 KB f32, O(T^2) combine never leaves the WGP) AND the
// first KC=96 rows of R (144 KB f16); only 160 rows/step stream from L2.
// Matvec: thread = (8-column block, 32-row k-chunk) -> b128 loads, 8 f32
// accumulators, kc-major LDS scratch reduction (bank-conflict-free).
// ---------------------------------------------------------------------------
__global__ void __launch_bounds__(H3, 1)
recurrence(const float* __restrict__ cond,    // [B, T, T]
           const _Float16* __restrict__ Rh,   // [H, 3H] f16
           const float* __restrict__ bias,    // [2, 3H]
           const float* __restrict__ MX,      // [B*T, 3H]
           float* __restrict__ out) {         // [B, T, H]
    __shared__ __align__(16) float    Yb[TT * HH];      // 128 KB history
    __shared__ __align__(16) _Float16 Rlds[KC * H3];    // 144 KB pinned R rows
    __shared__ __align__(16) float    scratch[8 * H3];  // 24 KB partials
    __shared__ float hprev[HH];
    __shared__ float cRow[TT];
    __shared__ float mhs[H3];

    const int b = blockIdx.x;
    const int t = threadIdx.x;       // 0..767
    const float* bias1 = bias + H3;

    // Pin R rows [0, KC) in LDS once (b128 copies)
    {
        const uint4* src = (const uint4*)Rh;
        uint4* dst = (uint4*)Rlds;
        for (int s = t; s < KC * H3 / 8; s += H3) dst[s] = src[s];
    }
    __syncthreads();

    const int nb = t % 96;           // 8-column block: cols [nb*8, nb*8+8)
    const int kc = t / 96;           // k-chunk: rows [kc*32, kc*32+32)

    for (int i = 0; i < TT; ++i) {
        // stage condition row cond[b, i, :]
        if (t < TT) cRow[t] = cond[(b * TT + i) * TT + t];
        __syncthreads();

        // (a) h_prev[h] = sum_{j<i} c[j] * Y_b[j][h]   (LDS-resident f32)
        {
            const int h = t & (HH - 1);
            const int c = t >> 8;                 // 3 j-stripes
            float acc = 0.f;
            for (int j = c; j < i; j += 3)
                acc += cRow[j] * Yb[j * HH + h];
            scratch[c * HH + h] = acc;
        }
        __syncthreads();
        if (t < HH) hprev[t] = scratch[t] + scratch[HH + t] + scratch[2 * HH + t];
        __syncthreads();

        // (b) partial mh: 8 cols x 32 k rows per thread; R from LDS (kc<3)
        // or streamed from L2 (kc>=3); v8h loads + 8 independent f32 accs.
        {
            float acc[8] = {0.f,0.f,0.f,0.f,0.f,0.f,0.f,0.f};
            const int k0 = kc * 32;
            if (kc < KC / 32) {
                const _Float16* rl = Rlds + k0 * H3 + nb * 8;
                #pragma unroll 4
                for (int kk = 0; kk < 32; ++kk) {
                    const float hk = hprev[k0 + kk];
                    v8h rv = *(const v8h*)(rl + kk * H3);
                    #pragma unroll
                    for (int c = 0; c < 8; ++c) acc[c] += hk * (float)rv[c];
                }
            } else {
                const _Float16* rg = Rh + k0 * H3 + nb * 8;
                #pragma unroll 4
                for (int kk = 0; kk < 32; ++kk) {
                    const float hk = hprev[k0 + kk];
                    v8h rv = *(const v8h*)(rg + kk * H3);
                    #pragma unroll
                    for (int c = 0; c < 8; ++c) acc[c] += hk * (float)rv[c];
                }
            }
            // kc-major scratch: 8 consecutive floats per thread, conflict-free
            float* sp = scratch + kc * H3 + nb * 8;
            #pragma unroll
            for (int c = 0; c < 8; ++c) sp[c] = acc[c];
        }
        __syncthreads();

        // reduce 8 k-chunks -> mh[n] (lane-contiguous reads)
        {
            float s = bias1[t];
            #pragma unroll
            for (int q = 0; q < 8; ++q) s += scratch[q * H3 + t];
            mhs[t] = s;
        }
        __syncthreads();

        // (c) gates + state update + output store
        if (t < HH) {
            const float* mx = MX + (b * TT + i) * H3;
            const float xz = mx[t];
            const float xr = mx[HH + t];
            const float xh = mx[2 * HH + t];
            const float z = 1.f / (1.f + __expf(-(xz + mhs[t])));
            const float r = 1.f / (1.f + __expf(-(xr + mhs[HH + t])));
            const float ct = xh + r * mhs[2 * HH + t];
            const float e2 = __expf(2.f * ct);            // tanh via one v_exp
            const float cand = (e2 - 1.f) / (e2 + 1.f);
            const float hn = z * hprev[t] + (1.f - z) * cand;
            Yb[i * HH + t] = hn;
            out[(b * TT + i) * HH + t] = hn;              // [B,T,H] direct
        }
        __syncthreads();
    }
}

// ---------------------------------------------------------------------------
// Host-side launch (graph-capture safe: only kernel launches on `stream`)
// ---------------------------------------------------------------------------
extern "C" void kernel_launch(void* const* d_in, const int* in_sizes, int n_in,
                              void* d_out, int out_size, void* d_ws, size_t ws_size,
                              hipStream_t stream) {
    const float* X    = (const float*)d_in[0];   // [B,T,D]
    const float* cond = (const float*)d_in[1];   // [B,T,T]
    const float* Kw   = (const float*)d_in[2];   // [D,3H]
    const float* Rw   = (const float*)d_in[3];   // [H,3H]
    const float* bias = (const float*)d_in[4];   // [2,3H]
    float* out        = (float*)d_out;           // [B,T,H]

    // Workspace layout (256B-aligned): Kt f16 | Rh f16 | MX f32
    char* ws = (char*)d_ws;
    _Float16* Kt = (_Float16*)(ws + 0);                       // 384 KB
    _Float16* Rh = (_Float16*)(ws + 393216);                  // 384 KB
    float*    MX = (float*)(ws + 786432);                     // 24 MB
    (void)in_sizes; (void)n_in; (void)out_size; (void)ws_size;

    // 1) f16 weight copies (kernel transposed for contiguous WMMA B-frags)
    convert_weights<<<(DD * H3 + 255) / 256, 256, 0, stream>>>(Kw, Rw, Kt, Rh);

    // 2) MX GEMM: 512 x 12 waves (4 tiles each), 8 waves/block -> 768 blocks
    mx_gemm_wmma<<<(BB * TT / 16) * (H3 / 64) / 8, 256, 0, stream>>>(X, Kt, bias, MX);

    // 3) Per-batch persistent recurrence: 64 WGs x 24 waves, ~300 KB LDS each
    recurrence<<<BB, H3, 0, stream>>>(cond, Rh, bias, MX, out);
}
